// MultiDomainNetNew_20873541058961
// MI455X (gfx1250) — compile-verified, matrix operand order checked by reading, not run on hardware
//
#include <hip/hip_runtime.h>

#define BB 8
#define CH 20
#define DD 64
#define LL 512
#define CC 2048
#define NF 12
#define K2 (2*CC)   // 4096 adjacency rows per domain

typedef __attribute__((ext_vector_type(16))) __bf16 v16bf;
typedef __attribute__((ext_vector_type(8)))  float  v8f;
typedef __attribute__((ext_vector_type(2)))  float  v2f;

__device__ __forceinline__ unsigned bf16bits(float f) {
  union { float f; unsigned u; } a; a.f = f;
  return (a.u + 0x7FFFu + ((a.u >> 16) & 1u)) >> 16;   // round-to-nearest-even
}
__device__ __forceinline__ unsigned pack2bf(float x, float y) {
  return (bf16bits(x) & 0xFFFFu) | (bf16bits(y) << 16);
}

// ---------------------------------------------------------------- k0: zero deg
__global__ void k0_zero(float* __restrict__ deg) {
  int i = blockIdx.x * blockDim.x + threadIdx.x;
  if (i < DD * LL) deg[i] = 0.f;
}

// ------------- k1: aai = aa_d @ adjs^T  (per-domain GEMM, bf16 WMMA, f32 acc),
//                   fused deg accumulation + stride-2 conv -> h workspace
__global__ __launch_bounds__(256)
void k1_gemm_conv(const float* __restrict__ aa, const float* __restrict__ adjs,
                  const float* __restrict__ conv_w, float* __restrict__ deg,
                  float* __restrict__ h) {
  __shared__ __align__(32) __bf16 Alds[160 * 32];   // 160 (b,ch) rows x 32 l
  __shared__ __align__(32) __bf16 Blds[64 * 32];    // 64 k cols (transposed) x 32 l
  __shared__ float degLds[32];
  __shared__ float aaiLds[160 * 64];                // full output tile for conv
  __shared__ float convLds[NF * CH * 2];

  const int tid  = threadIdx.x;
  const int wave = tid >> 5;
  const int lane = tid & 31;
  const int d     = blockIdx.x >> 6;                 // domain
  const int kbase = (blockIdx.x & 63) * 64;          // N-offset into 4096 k rows

  // wave -> fixed N column: B fragment loaded once per K-chunk
  const int nt     = wave & 3;
  const int mtBase = (wave >> 2) * 5;                // 5 M tiles per wave (10 total)

  unsigned* AldsU = (unsigned*)Alds;
  unsigned* BldsU = (unsigned*)Blds;

  for (int e = tid; e < NF * CH * 2; e += 256) convLds[e] = conv_w[e];
  if (tid < 32) degLds[tid] = 0.f;

  v8f acc[5];
  v8f z = {};
  #pragma unroll
  for (int i = 0; i < 5; ++i) acc[i] = z;

  const size_t adjBase = (size_t)d * K2 * LL;
  const int aOff = (mtBase * 16 + (lane & 15)) * 32 + (lane >> 4) * 16;
  const int bOff = (nt * 16 + (lane & 15)) * 32 + (lane >> 4) * 16;
  __syncthreads();

  for (int lb = 0; lb < LL; lb += 32) {
    // stage A: aa rows, 2 elems/thread-iter (global b64 -> one packed LDS b32)
    for (int e = tid; e < 160 * 16; e += 256) {
      int r = e >> 4, lc2 = (e & 15) * 2;
      v2f v = *(const v2f*)&aa[(size_t)r * (DD * LL) + (size_t)d * LL + lb + lc2];
      AldsU[e] = pack2bf(v.x, v.y);
    }
    // stage B transposed [n][lc], non-temporal stream + fused degree partials
    float dp0 = 0.f, dp1 = 0.f;
    for (int e = tid; e < 64 * 16; e += 256) {
      int n = e >> 4, lc2 = (e & 15) * 2;
      v2f v = __builtin_nontemporal_load(
          (const v2f*)&adjs[adjBase + (size_t)(kbase + n) * LL + lb + lc2]);
      BldsU[n * 16 + (e & 15)] = pack2bf(v.x, v.y);
      dp0 += v.x; dp1 += v.y;                 // lc2 invariant across this loop
    }
    atomicAdd(&degLds[(tid & 15) * 2    ], dp0);
    atomicAdd(&degLds[(tid & 15) * 2 + 1], dp1);
    if (lb + 32 < LL)
      __builtin_prefetch(&adjs[adjBase + (size_t)kbase * LL + lb + 32], 0, 1);
    __syncthreads();

    v16bf bv = *(const v16bf*)&Blds[bOff];    // one B fragment per wave per chunk
    #pragma unroll
    for (int i = 0; i < 5; ++i) {
      v16bf av = *(const v16bf*)&Alds[aOff + i * 16 * 32];
      acc[i] = __builtin_amdgcn_wmma_f32_16x16x32_bf16(false, av, false, bv,
                                                       (short)0, acc[i], false, false);
    }
    __syncthreads();
    if (tid < 32) {                           // flush degree partial for l-chunk
      atomicAdd(&deg[d * LL + lb + tid], degLds[tid]);
      degLds[tid] = 0.f;
    }
    __syncthreads();
  }

  // spill accumulators to LDS per WMMA C-layout (vgpr v -> row m = v + 8*laneHi)
  {
    int hi8 = (lane >= 16) ? 8 : 0;
    int col = nt * 16 + (lane & 15);
    #pragma unroll
    for (int i = 0; i < 5; ++i) {
      int mt = mtBase + i;
      #pragma unroll
      for (int v = 0; v < 8; ++v)
        aaiLds[(mt * 16 + v + hi8) * 64 + col] = acc[i][v];
    }
  }
  __syncthreads();

  // stride-2 conv over adjacent k columns, contracting CH -> 96 rows x 32 pairs
  for (int e = tid; e < 96 * 32; e += 256) {
    int r = e >> 5, p = e & 31;
    int b = r / NF, f = r % NF;
    float s = 0.f;
    #pragma unroll
    for (int c = 0; c < CH; ++c) {
      s += aaiLds[(b * CH + c) * 64 + 2 * p    ] * convLds[(f * CH + c) * 2    ];
      s += aaiLds[(b * CH + c) * 64 + 2 * p + 1] * convLds[(f * CH + c) * 2 + 1];
    }
    h[((size_t)r * DD + d) * CC + (kbase >> 1) + p] = s;
  }
}

// ------------- k2: aa2 = h @ adjs_even (bf16 WMMA), /deg, row-max over 64 l's
__global__ __launch_bounds__(256)
void k2_gemm_max(const float* __restrict__ adjs, const float* __restrict__ h,
                 const float* __restrict__ deg, float* __restrict__ m2) {
  __shared__ __align__(32) __bf16 Alds[96 * 32];   // 96 (b,f) rows x 32 p
  __shared__ __align__(32) __bf16 Blds[64 * 32];   // 64 l cols (transposed) x 32 p
  __shared__ float accLds[96 * 64];
  __shared__ float degT[64];

  const int tid  = threadIdx.x;
  const int wave = tid >> 5;
  const int lane = tid & 31;
  const int d     = blockIdx.x >> 3;
  const int lt    = blockIdx.x & 7;
  const int lbase = lt * 64;

  const int nt     = wave & 3;
  const int mtBase = (wave >> 2) * 3;               // 3 M tiles per wave (6 total)

  unsigned* AldsU = (unsigned*)Alds;
  unsigned short* BldsS = (unsigned short*)Blds;

  if (tid < 64) degT[tid] = deg[d * LL + lbase + tid];

  v8f acc[3];
  v8f z = {};
  #pragma unroll
  for (int i = 0; i < 3; ++i) acc[i] = z;

  const size_t adjBase = (size_t)d * K2 * LL;
  const int aOff = (mtBase * 16 + (lane & 15)) * 32 + (lane >> 4) * 16;
  const int bOff = (nt * 16 + (lane & 15)) * 32 + (lane >> 4) * 16;
  __syncthreads();

  for (int pb = 0; pb < CC; pb += 32) {
    for (int e = tid; e < 96 * 16; e += 256) {       // A: h rows, paired
      int r = e >> 4, pc2 = (e & 15) * 2;
      v2f v = *(const v2f*)&h[((size_t)r * DD + d) * CC + pb + pc2];
      AldsU[e] = pack2bf(v.x, v.y);
    }
    for (int e = tid; e < 64 * 32; e += 256) {       // B: even k rows, stream
      int n = e & 63, pc = e >> 6;                   // coalesced over l (n)
      float v = __builtin_nontemporal_load(
          &adjs[adjBase + (size_t)(2 * (pb + pc)) * LL + lbase + n]);
      BldsS[n * 32 + pc] = (unsigned short)bf16bits(v);
    }
    __syncthreads();

    v16bf bv = *(const v16bf*)&Blds[bOff];
    #pragma unroll
    for (int i = 0; i < 3; ++i) {
      v16bf av = *(const v16bf*)&Alds[aOff + i * 16 * 32];
      acc[i] = __builtin_amdgcn_wmma_f32_16x16x32_bf16(false, av, false, bv,
                                                       (short)0, acc[i], false, false);
    }
    __syncthreads();
  }

  {
    int hi8 = (lane >= 16) ? 8 : 0;
    int col = nt * 16 + (lane & 15);
    #pragma unroll
    for (int i = 0; i < 3; ++i) {
      int mt = mtBase + i;
      #pragma unroll
      for (int v = 0; v < 8; ++v)
        accLds[(mt * 16 + v + hi8) * 64 + col] = acc[i][v] / degT[col];
    }
  }
  __syncthreads();

  if (tid < 96) {   // exclusive (row, d, lt) slot -> plain store, no atomics
    float m = -3.402823466e38f;
    for (int c = 0; c < 64; ++c) m = fmaxf(m, accLds[tid * 64 + c]);
    m2[((size_t)tid * DD + d) * 8 + lt] = m;
  }
}

// ------------- k3: per (d,b) raw-channel max over L + combine with lin_w
__global__ __launch_bounds__(256)
void k3_final(const float* __restrict__ aa, const float* __restrict__ m2,
              const float* __restrict__ lin, float* __restrict__ out) {
  __shared__ float red[256];
  __shared__ float score;
  const int tid = threadIdx.x;
  const int d = blockIdx.x / BB;
  const int b = blockIdx.x % BB;
  if (tid == 0) score = 0.f;
  __syncthreads();
  for (int ch = 0; ch < CH; ++ch) {
    float m = -3.402823466e38f;
    for (int l = tid; l < LL; l += 256)
      m = fmaxf(m, aa[(size_t)(b * CH + ch) * (DD * LL) + (size_t)d * LL + l]);
    red[tid] = m;
    __syncthreads();
    for (int s = 128; s > 0; s >>= 1) {
      if (tid < s) red[tid] = fmaxf(red[tid], red[tid + s]);
      __syncthreads();
    }
    if (tid == 0) score += red[0] * lin[ch];
    __syncthreads();
  }
  if (tid == 0) {
    float sc = score;
    for (int f = 0; f < NF; ++f) {
      float m = -3.402823466e38f;
      for (int t = 0; t < 8; ++t)
        m = fmaxf(m, m2[((size_t)(b * NF + f) * DD + d) * 8 + t]);
      sc += m * lin[CH + f];
    }
    out[d * BB + b] = sc;
  }
}

// ---------------------------------------------------------------- launch
extern "C" void kernel_launch(void* const* d_in, const int* in_sizes, int n_in,
                              void* d_out, int out_size, void* d_ws, size_t ws_size,
                              hipStream_t stream) {
  (void)in_sizes; (void)n_in; (void)out_size; (void)ws_size;
  const float* aa     = (const float*)d_in[0];
  const float* adjs   = (const float*)d_in[1];
  const float* conv_w = (const float*)d_in[2];
  const float* lin_w  = (const float*)d_in[3];
  float* out = (float*)d_out;

  float* ws  = (float*)d_ws;
  float* deg = ws;                                      // D*L         = 32768 f
  float* m2  = ws + DD * LL;                            // (B*NF)*D*8  = 49152 f
  float* h   = ws + DD * LL + (size_t)BB * NF * DD * 8; // (B*NF)*D*C  = 12.6M f

  k0_zero<<<(DD * LL + 255) / 256, 256, 0, stream>>>(deg);
  k1_gemm_conv<<<DD * 64, 256, 0, stream>>>(aa, adjs, conv_w, deg, h);
  k2_gemm_max<<<DD * 8, 256, 0, stream>>>(adjs, h, deg, m2);
  k3_final<<<DD * BB, 256, 0, stream>>>(aa, m2, lin_w, out);
}
